// BandSplitModule_28879360098326
// MI455X (gfx1250) — matrix operand order repaired and checked by reading, not compiled, not run on hardware
//
#include <hip/hip_runtime.h>
#include <hip/hip_bf16.h>

// ---------------------------------------------------------------------------
// BandSplit: per-band LayerNorm([C,T]) + Linear(C->128), 41 bands, B=4, T=517.
// Memory-bound: ~255 MB compulsory HBM traffic -> ~11 us floor @ 23.3 TB/s.
// Three passes (x fits in the 192 MB L2, so re-reads are cheap):
//   pass 0: W (f32, [C,128]) -> f16 pre-swizzled into WMMA B-fragment order,
//           K-tail zero-padded (removes bounds select + 16x VMEM in hot loop)
//   pass 1: per-(band,batch) mean / rsqrt(var+eps), f32 partials, LDS reduce
//   pass 2: fused normalize -> f16 -> v_wmma_f32_16x16x32_f16 -> +bias
// ---------------------------------------------------------------------------

typedef _Float16 v16h __attribute__((ext_vector_type(16)));
typedef float    v8f  __attribute__((ext_vector_type(8)));

#define N_BANDS  41
#define T_STEPS  517
#define N_FREQ   1025
#define FC_DIM   128
#define BATCH    4
#define MT       2            // t-tiles (of 16) per block
#define N_TGROUP 17           // ceil(517 / (16*MT))
#define MAXCHUNK 48           // max ceil(C/32) over bands (bw=96)

// Band (start,end) freq-bin indices from the reference _band_indices().
__constant__ int BAND_S[N_BANDS] = {
    0, 5, 10, 14, 19, 24, 28, 33, 38, 42,
    47, 59, 70, 82, 93, 105, 117, 128, 140, 151, 163, 175,
    186, 209, 233, 256, 279, 302, 326, 349,
    372, 418, 465, 511, 558, 604, 651, 697,
    744, 836, 929};
__constant__ int BAND_E[N_BANDS] = {
    5, 10, 14, 19, 24, 28, 33, 38, 42, 47,
    59, 70, 82, 93, 105, 117, 128, 140, 151, 163, 175, 186,
    209, 233, 256, 279, 302, 326, 349, 372,
    418, 465, 511, 558, 604, 651, 697, 744,
    836, 929, 1025};
// Exact div-by-bw: fm = (cg * MAGIC) >> 22, MAGIC = ceil(2^22/bw).
// Exact for cg < 1536, bw <= 96 (error < 1535/2^22 < 1/96).
__constant__ unsigned BAND_MAGIC[N_BANDS] = {
    838861, 838861, 1048576, 838861, 838861, 1048576, 838861, 838861, 1048576, 838861,
    349526, 381301, 349526, 381301, 349526, 349526, 381301, 349526, 381301, 349526, 349526, 381301,
    182362, 174763, 182362, 182362, 182362, 174763, 182362, 182362,
    91181, 89241, 91181, 89241, 91181, 89241, 91181, 89241,
    45591, 45101, 43691};
// Prefix sum of nchunk = ceil(bw/2) per band (for swizzled-W offsets).
__constant__ int BAND_CUM[N_BANDS] = {
    0, 3, 6, 8, 11, 14, 16, 19, 22, 24,
    27, 33, 39, 45, 51, 57, 63, 69, 75, 81, 87, 93,
    99, 111, 123, 135, 147, 159, 171, 183,
    195, 218, 242, 265, 289, 312, 336, 359,
    383, 429, 476};     // total 524 fragments-groups; Wf16 = 524*4096 halfs

struct BandPtrs {
    const float* g[N_BANDS];     // [C,T] layernorm gamma
    const float* be[N_BANDS];    // [C,T] layernorm beta
    const float* w[N_BANDS];     // [C,128] linear weight (in-major)
    const float* bias[N_BANDS];  // [128]
};

// ---------------------------------------------------------------------------
// Pass 0: swizzle W -> f16 B fragments.
// Fragment (band, dtile, kchunk): 32 lanes x 16 halfs, B 32x16 in contiguous-K order (same as A staging):
//   lane<16 holds K=0..15 (half j = K), lane>=16 holds K=16..31; col = lane&15.
// grid = (41, 8, MAXCHUNK), block = 256; kc >= nchunk blocks exit.
// ---------------------------------------------------------------------------
__launch_bounds__(256)
__global__ void bs_wswz(BandPtrs bp, _Float16* __restrict__ wf16) {
    const int band  = blockIdx.x;
    const int dtile = blockIdx.y;
    const int kc    = blockIdx.z;
    const int C      = (BAND_E[band] - BAND_S[band]) * 16;
    const int nchunk = (C + 31) >> 5;
    if (kc >= nchunk) return;

    const float* __restrict__ w = bp.w[band];
    const size_t fragBase =
        ((size_t)BAND_CUM[band] * 8 + (size_t)dtile * nchunk + kc) * 512;

#pragma unroll
    for (int r = 0; r < 2; ++r) {
        const int flat = threadIdx.x + r * 256;  // 0..511 = lane*16 + j
        const int lane = flat >> 4;
        const int j    = flat & 15;
        const int cg   = (kc << 5) + ((lane >> 4) << 4) + j;
        float v = 0.0f;
        if (cg < C) v = w[(size_t)cg * FC_DIM + dtile * 16 + (lane & 15)];
        wf16[fragBase + flat] = (_Float16)v;     // contiguous b16 stores
    }
}

// ---------------------------------------------------------------------------
// Pass 1: per-(band,batch) sum / sumsq.  x slab for fixed (b,fm) is contiguous.
// ---------------------------------------------------------------------------
__launch_bounds__(256)
__global__ void bs_stats(const float* __restrict__ x, float2* __restrict__ stats) {
    const int band = blockIdx.x % N_BANDS;
    const int b    = blockIdx.x / N_BANDS;
    const int s    = BAND_S[band];
    const int span = (BAND_E[band] - s) * T_STEPS;

    float s1 = 0.0f, s2 = 0.0f;
    for (int fm = 0; fm < 16; ++fm) {
        const float* p = x + ((size_t)(b * 16 + fm) * N_FREQ + s) * T_STEPS;
        for (int j = threadIdx.x; j < span; j += 256) {
            float v = p[j];
            s1 += v;
            s2 += v * v;
        }
    }

    __shared__ float r1[256];
    __shared__ float r2[256];
    r1[threadIdx.x] = s1;
    r2[threadIdx.x] = s2;
    __syncthreads();
    for (int off = 128; off > 0; off >>= 1) {
        if (threadIdx.x < off) {
            r1[threadIdx.x] += r1[threadIdx.x + off];
            r2[threadIdx.x] += r2[threadIdx.x + off];
        }
        __syncthreads();
    }
    if (threadIdx.x == 0) {
        float n   = (float)span * 16.0f;
        float mu  = r1[0] / n;
        float var = r2[0] / n - mu * mu;
        stats[blockIdx.x] = make_float2(mu, rsqrtf(var + 1e-5f));
    }
}

// ---------------------------------------------------------------------------
// Pass 2: one block (8 wave32) per (band, t-group of 32 rows, batch).
// Wave w owns output columns [w*16, w*16+16); one B frag feeds MT=2 WMMAs.
// ---------------------------------------------------------------------------
__launch_bounds__(256)
__global__ void bs_gemm(const float* __restrict__ x, BandPtrs bp,
                        const float2* __restrict__ stats,
                        const v16h* __restrict__ wf16,
                        float* __restrict__ out) {
    const int band   = blockIdx.x / N_TGROUP;
    const int tgrp   = blockIdx.x % N_TGROUP;
    const int b      = blockIdx.y;
    const int s      = BAND_S[band];
    const int bw     = BAND_E[band] - s;
    const int C      = bw * 16;
    const int nchunk = (C + 31) >> 5;
    const unsigned magic = BAND_MAGIC[band];
    const int t0     = tgrp * (16 * MT);

    const float2 st  = stats[b * N_BANDS + band];
    const float  mu  = st.x;
    const float  inv = st.y;

    const float* __restrict__ gptr  = bp.g[band];
    const float* __restrict__ beptr = bp.be[band];

    const int tid   = threadIdx.x;
    const int lane  = tid & 31;           // wave32
    const int wave  = tid >> 5;           // 0..7 -> dbase = wave*16
    const int ncol  = lane & 15;

    // Per-wave swizzled-W fragment stream (contiguous 1 KB per chunk).
    const v16h* __restrict__ bfragp =
        wf16 + ((size_t)BAND_CUM[band] * 8 + (size_t)wave * nchunk) * 32 + lane;

    // A fragments staged in LDS as f16 16x32, contiguous K (k = 16*(lane/16) + j), same as B.
    __shared__ v16h aTileV[MT * 32];
    _Float16* aH = reinterpret_cast<_Float16*>(aTileV);

    v8f acc0 = {};
    v8f acc1 = {};
    for (int kc = 0; kc < nchunk; ++kc) {
        const int cbase = kc << 5;

        // --- stage MT*512 normalized f16 elements, 4 per thread ---
#pragma unroll
        for (int r = 0; r < 2 * MT; ++r) {
            const int flat = tid + r * 256;        // 0..1023
            const int tile = flat >> 9;            // 0..MT-1
            const int rem  = flat & 511;           // klocal*16 + tlocal
            const int kl   = rem >> 4;
            const int tl   = rem & 15;
            const int cg   = cbase + kl;
            const int tg   = t0 + tile * 16 + tl;
            float val = 0.0f;
            if (cg < C && tg < T_STEPS) {
                const int fm = (int)(((unsigned)cg * magic) >> 22); // exact cg/bw
                const int fl = cg - fm * bw;
                const float xv = x[((size_t)(b * 16 + fm) * N_FREQ + (s + fl)) * T_STEPS + tg];
                const float gv = gptr[(size_t)cg * T_STEPS + tg];
                const float bv = beptr[(size_t)cg * T_STEPS + tg];
                val = (xv - mu) * inv * gv + bv;
            }
            const int dlane = tl + ((kl & 16) ? 16 : 0);
            const int dj    = kl & 15;
            aH[tile * 512 + dlane * 16 + dj] = (_Float16)val;
        }
        __syncthreads();

        const v16h bf = bfragp[(size_t)kc * 32];   // 2x global_load_b128
        const v16h a0 = aTileV[lane];              // ds_load_b128 x2 (broadcast)
        const v16h a1 = aTileV[32 + lane];

        acc0 = __builtin_amdgcn_wmma_f32_16x16x32_f16(false, a0, false, bf,
                                                      (short)0, acc0, false, false);
        acc1 = __builtin_amdgcn_wmma_f32_16x16x32_f16(false, a1, false, bf,
                                                      (short)0, acc1, false, false);
        __syncthreads();   // protect aTileV before next chunk overwrite
    }

    // --- epilogue: C/D layout (VGPR r -> row r + 8*(lane>=16), col = lane&15) ---
    const float bias = bp.bias[band][wave * 16 + ncol];
    const int mrow = ((lane >> 4) << 3);
#pragma unroll
    for (int r = 0; r < 8; ++r) {
        const int t = t0 + r + mrow;
        if (t < T_STEPS)
            out[(((size_t)b * N_BANDS + band) * T_STEPS + t) * FC_DIM + wave * 16 + ncol] =
                acc0[r] + bias;
    }
#pragma unroll
    for (int r = 0; r < 8; ++r) {
        const int t = t0 + 16 + r + mrow;
        if (t < T_STEPS)
            out[(((size_t)b * N_BANDS + band) * T_STEPS + t) * FC_DIM + wave * 16 + ncol] =
                acc1[r] + bias;
    }
}

// ---------------------------------------------------------------------------
extern "C" void kernel_launch(void* const* d_in, const int* in_sizes, int n_in,
                              void* d_out, int out_size, void* d_ws, size_t ws_size,
                              hipStream_t stream) {
    (void)in_sizes; (void)n_in; (void)out_size; (void)ws_size;
    const float* x = (const float*)d_in[0];
    BandPtrs bp;
    for (int i = 0; i < N_BANDS; ++i) {
        bp.g[i]    = (const float*)d_in[1 + i];                // ln_gammas
        bp.be[i]   = (const float*)d_in[1 + N_BANDS + i];      // ln_betas
        bp.w[i]    = (const float*)d_in[1 + 2 * N_BANDS + i];  // fc_ws
        bp.bias[i] = (const float*)d_in[1 + 3 * N_BANDS + i];  // fc_bs
    }
    float2*   stats = (float2*)d_ws;                           // 164 * 8 B
    _Float16* wf16  = (_Float16*)((char*)d_ws + 2048);         // 524*4096 halfs (~4.3 MB)
    float*    out   = (float*)d_out;

    bs_wswz<<<dim3(N_BANDS, 8, MAXCHUNK), dim3(256), 0, stream>>>(bp, wf16);
    bs_stats<<<dim3(N_BANDS * BATCH), dim3(256), 0, stream>>>(x, stats);
    bs_gemm<<<dim3(N_BANDS * N_TGROUP, BATCH), dim3(256), 0, stream>>>(
        x, bp, stats, (const v16h*)wf16, out);
}
